// FMT_with_CVPE_61546881351875
// MI455X (gfx1250) — compile-verified, matrix operand order checked by reading, not run on hardware
//
#include <hip/hip_runtime.h>
#include <hip/hip_bf16.h>
#include <math.h>

// ---------------- problem constants ----------------
#define HH_   128
#define WW_   160
#define HWP   20480            // H*W
#define LTOK  20480            // tokens per batch
#define MTOK  40960            // 2 batches of tokens

typedef __bf16 v16bf __attribute__((ext_vector_type(16)));
typedef float  v8f   __attribute__((ext_vector_type(8)));

union BF16U { __bf16 b; unsigned short u; };
__device__ __forceinline__ unsigned short f2bfu(float f) { BF16U c; c.b = (__bf16)f; return c.u; }

union FRAG { v16bf v; uint4 q[2]; };

// =====================================================================
// Weight pre-pack: f32 [K][N] row-major -> bf16 fragments laid out as
//   packed[kstep][n16][lane][8 x u32]   (u32 = bf16 pair, low = even K)
// so each lane's 16x16x32 B-fragment is 32 contiguous bytes.
// K mapping per ISA 16-bit matrix layout:
//   lane<16 : K = {2j,2j+1}(j<4), {16+2j',...}(j>=4); lane>=16: +8
// =====================================================================
__global__ __launch_bounds__(256) void pack_b_kernel(
    const float* __restrict__ Bin, unsigned int* __restrict__ Bout, int K, int N)
{
  const float* Bm = Bin + (size_t)blockIdx.y * K * N;
  unsigned int* Bo = Bout + (size_t)blockIdx.y * ((size_t)K * N / 2);
  int total = K * N / 2;
  int idx = blockIdx.x * 256 + threadIdx.x;
  if (idx >= total) return;
  int nTiles = N >> 4;
  int perK = nTiles << 8;            // u32 per k-step
  int kstep = idx / perK;
  int rem = idx % perK;
  int n16 = rem >> 8;
  int rem2 = rem & 255;
  int lane = rem2 >> 3;
  int j = rem2 & 7;
  int k = kstep * 32 + ((j < 4) ? (2 * j) : (16 + 2 * (j - 4))) + (lane >> 4) * 8;
  int n = n16 * 16 + (lane & 15);
  unsigned int lo = f2bfu(Bm[(size_t)k * N + n]);
  unsigned int hi = f2bfu(Bm[(size_t)(k + 1) * N + n]);
  Bo[idx] = lo | (hi << 16);
}

// A fragment: two contiguous 8-float runs -> 4 x b128 loads + cvt
__device__ __forceinline__ v16bf make_afrag(const float* __restrict__ row, int off)
{
  const float4* p0 = reinterpret_cast<const float4*>(row + off);
  const float4* p1 = reinterpret_cast<const float4*>(row + off + 16);
  float4 u0 = p0[0], u1 = p0[1], u2 = p1[0], u3 = p1[1];
  v16bf a;
  a[0]  = (__bf16)u0.x; a[1]  = (__bf16)u0.y; a[2]  = (__bf16)u0.z; a[3]  = (__bf16)u0.w;
  a[4]  = (__bf16)u1.x; a[5]  = (__bf16)u1.y; a[6]  = (__bf16)u1.z; a[7]  = (__bf16)u1.w;
  a[8]  = (__bf16)u2.x; a[9]  = (__bf16)u2.y; a[10] = (__bf16)u2.z; a[11] = (__bf16)u2.w;
  a[12] = (__bf16)u3.x; a[13] = (__bf16)u3.y; a[14] = (__bf16)u3.z; a[15] = (__bf16)u3.w;
  return a;
}

// =====================================================================
// bf16 WMMA GEMM: out[M,N] = act( A[M,K] @ B[K,N] + bias (+ resid) )
// Block 256 thr = 8 waves; block tile 256(M) x 64(N); wave tile 32x64.
// B pre-packed in fragment order; staged to LDS with b128 copies;
// fragment read = 2 x ds_load_b128. 8 WMMA per K-step per wave.
// act: 0 none, 1 relu, 2 elu(x)+1
// =====================================================================
__global__ __launch_bounds__(256) void gemm_bf16_kernel(
    const float* __restrict__ A,
    const unsigned int* __restrict__ Bp,
    const float* __restrict__ bias,
    const float* __restrict__ resid,
    float* __restrict__ out,
    int M, int N, int K, int act)
{
  __shared__ uint4 sB4[2048];                 // up to (K=256/32)*256 uint4 = 32KB
  const int tid = threadIdx.x;
  const int m0 = blockIdx.x * 256;
  const int n0 = blockIdx.y * 64;
  const int nTiles = N >> 4;
  const int n16b = n0 >> 4;
  const int nq = (K >> 5) << 8;               // uint4 count for this block
  const uint4* Bq = reinterpret_cast<const uint4*>(Bp);

  for (int i = tid; i < nq; i += 256) {
    int kstep = i >> 8;
    int rem = i & 255;                        // 4 tiles x 64 uint4, contiguous
    sB4[i] = Bq[(size_t)(kstep * nTiles + n16b) * 64 + rem];
  }
  __syncthreads();

  const int wave  = tid >> 5;
  const int lane  = tid & 31;
  const int lmod  = lane & 15;
  const int lhalf = lane >> 4;
  const int khalf = lhalf * 8;

  v8f acc[2][4];
#pragma unroll
  for (int r = 0; r < 2; ++r)
#pragma unroll
    for (int nt = 0; nt < 4; ++nt) acc[r][nt] = (v8f){};

  const float* A0 = A + (size_t)(m0 + wave * 32 + lmod) * K;
  const float* A1 = A0 + (size_t)16 * K;

  int kstep = 0;
  for (int k0 = 0; k0 < K; k0 += 32, ++kstep) {
    v16bf af0 = make_afrag(A0, k0 + khalf);
    v16bf af1 = make_afrag(A1, k0 + khalf);
#pragma unroll
    for (int nt = 0; nt < 4; ++nt) {
      FRAG fb;
      int qi = kstep * 256 + nt * 64 + lane * 2;
      fb.q[0] = sB4[qi];
      fb.q[1] = sB4[qi + 1];
      acc[0][nt] = __builtin_amdgcn_wmma_f32_16x16x32_bf16(
          false, af0, false, fb.v, (short)0, acc[0][nt], false, false);
      acc[1][nt] = __builtin_amdgcn_wmma_f32_16x16x32_bf16(
          false, af1, false, fb.v, (short)0, acc[1][nt], false, false);
    }
  }

#pragma unroll
  for (int r = 0; r < 2; ++r)
#pragma unroll
    for (int nt = 0; nt < 4; ++nt) {
      int ncol = n0 + nt * 16 + lmod;
      float bv = bias[ncol];
#pragma unroll
      for (int i = 0; i < 8; ++i) {
        int mm = m0 + wave * 32 + r * 16 + lhalf * 8 + i;   // C/D layout
        size_t o = (size_t)mm * N + ncol;
        float val = acc[r][nt][i] + bv;
        if (resid) val += resid[o];
        if (act == 1)      val = fmaxf(val, 0.0f);
        else if (act == 2) val = (val > 0.0f) ? (val + 1.0f) : expf(val);
        out[o] = val;
      }
    }
}

__global__ __launch_bounds__(256) void zero_kernel(float* __restrict__ p, int n)
{
  int i = blockIdx.x * 256 + threadIdx.x;
  if (i < n) p[i] = 0.0f;
}

// ---------------- SE gate (tiny MLP chain), one block ----------------
__global__ __launch_bounds__(256) void cam_se_kernel(
    const float* __restrict__ cam16,
    const float* __restrict__ bn1_g, const float* __restrict__ bn1_b,
    const float* __restrict__ fc1_w, const float* __restrict__ fc1_b,
    const float* __restrict__ fc2_w, const float* __restrict__ fc2_b,
    const float* __restrict__ ser_w, const float* __restrict__ ser_b,
    const float* __restrict__ see_w, const float* __restrict__ see_b,
    float* __restrict__ gate)
{
  __shared__ float s_in[2][16];
  __shared__ float s_a[2][128];
  __shared__ float s_b[2][128];
  const int tid = threadIdx.x;
  const int b = tid >> 7;
  const int o = tid & 127;
  const float inv_std = 0.9999950000374996f;    // 1/sqrt(1+1e-5)

  if (tid < 32) {
    int bb = tid >> 4, j = tid & 15;
    s_in[bb][j] = cam16[bb * 16 + j] * inv_std * bn1_g[j] + bn1_b[j];
  }
  __syncthreads();

  float a = fc1_b[o];
  for (int j = 0; j < 16; ++j) a += s_in[b][j] * fc1_w[j * 128 + o];
  s_a[b][o] = fmaxf(a, 0.0f);
  __syncthreads();

  float se = fc2_b[o];
  for (int j = 0; j < 128; ++j) se += s_a[b][j] * fc2_w[j * 128 + o];
  s_b[b][o] = se;
  __syncthreads();

  float g1 = ser_b[o];
  for (int j = 0; j < 128; ++j) g1 += s_b[b][j] * ser_w[j * 128 + o];
  s_a[b][o] = fmaxf(g1, 0.0f);
  __syncthreads();

  float g2 = see_b[o];
  for (int j = 0; j < 128; ++j) g2 += s_a[b][j] * see_w[j * 128 + o];
  gate[b * 128 + o] = 1.0f / (1.0f + expf(-g2));
}

// ---------------- conv3x3(depth) + BN + relu + gate -> ctx[pix][c] ----------------
__global__ __launch_bounds__(256) void ctx_kernel(
    const float* __restrict__ depth, const float* __restrict__ rc_w,
    const float* __restrict__ rc_b, const float* __restrict__ bn2_g,
    const float* __restrict__ bn2_b, const float* __restrict__ gate,
    float* __restrict__ ctx)
{
  int p = blockIdx.x * 256 + threadIdx.x;
  if (p >= MTOK) return;
  int b = p / HWP, pp = p % HWP;
  int y = pp / WW_, x = pp % WW_;
  float nb[9];
#pragma unroll
  for (int dy = 0; dy < 3; ++dy)
#pragma unroll
    for (int dx = 0; dx < 3; ++dx) {
      int yy = y + dy - 1, xx = x + dx - 1;
      bool ok = (yy >= 0) && (yy < HH_) && (xx >= 0) && (xx < WW_);
      nb[dy * 3 + dx] = ok ? depth[(size_t)b * HWP + yy * WW_ + xx] : 0.0f;
    }
  const float inv_std = 0.9999950000374996f;
  for (int c = 0; c < 128; c += 4) {
    float4 o4;
    float* op = reinterpret_cast<float*>(&o4);
#pragma unroll
    for (int u = 0; u < 4; ++u) {
      int cc = c + u;
      float s = rc_b[cc];
#pragma unroll
      for (int i = 0; i < 9; ++i) s += rc_w[cc * 9 + i] * nb[i];
      s = s * inv_std * bn2_g[cc] + bn2_b[cc];
      s = fmaxf(s, 0.0f) * gate[b * 128 + cc];
      op[u] = s;
    }
    *reinterpret_cast<float4*>(ctx + (size_t)p * 128 + c) = o4;
  }
}

// ---------------- homography warp ref_ctx -> src_ctx (bilinear, zero pad) ----------------
__global__ __launch_bounds__(256) void warp_kernel(
    const float* __restrict__ refctx, const float* __restrict__ depth,
    const float* __restrict__ ref_proj, const float* __restrict__ src_i2w,
    float* __restrict__ srcctx)
{
  int pid = blockIdx.x * 2 + (threadIdx.x >> 7);
  int c = threadIdx.x & 127;
  if (pid >= MTOK) return;
  int b = pid / HWP, pp = pid % HWP;
  int y = pp / WW_, x = pp % WW_;

  float P[12];
  const float* Rp = ref_proj + b * 16;
  const float* Si = src_i2w + b * 16;
#pragma unroll
  for (int i = 0; i < 3; ++i)
#pragma unroll
    for (int j = 0; j < 4; ++j) {
      float s = 0.0f;
#pragma unroll
      for (int kk = 0; kk < 4; ++kk) s += Rp[i * 4 + kk] * Si[kk * 4 + j];
      P[i * 4 + j] = s;
    }
  float dep = depth[b * HWP + pp];
  float fx = (float)x, fy = (float)y;
  float px = (P[0] * fx + P[1] * fy + P[2]) * dep + P[3];
  float py = (P[4] * fx + P[5] * fy + P[6]) * dep + P[7];
  float pz = (P[8] * fx + P[9] * fy + P[10]) * dep + P[11];
  if (pz == 0.0f) pz = 1e-9f;
  float sx = px / pz, sy = py / pz;
  float x0 = floorf(sx), y0 = floorf(sy);
  float wx = sx - x0, wy = sy - y0;

  float val = 0.0f;
#pragma unroll
  for (int cy = 0; cy < 2; ++cy)
#pragma unroll
    for (int cx = 0; cx < 2; ++cx) {
      float xi = x0 + (float)cx, yi = y0 + (float)cy;
      bool valid = (xi >= 0.0f) && (xi <= (float)(WW_ - 1)) &&
                   (yi >= 0.0f) && (yi <= (float)(HH_ - 1));
      int xc = (int)fminf(fmaxf(xi, 0.0f), (float)(WW_ - 1));
      int yc = (int)fminf(fmaxf(yi, 0.0f), (float)(HH_ - 1));
      float w = (cx ? wx : 1.0f - wx) * (cy ? wy : 1.0f - wy);
      float vv = valid ? refctx[((size_t)(b * HWP) + yc * WW_ + xc) * 128 + c] : 0.0f;
      val += w * vv;
    }
  srcctx[(size_t)pid * 128 + c] = val;
}

// ---------------- tokens = feature(transposed) + sine PE + ctx ----------------
__global__ __launch_bounds__(256) void assemble_kernel(
    const float* __restrict__ refF, const float* __restrict__ srcF,
    const float* __restrict__ refctx, const float* __restrict__ srcctx,
    float* __restrict__ Xref, float* __restrict__ Xsrc)
{
  size_t idx = (size_t)blockIdx.x * 256 + threadIdx.x;
  if (idx >= (size_t)MTOK * 128) return;
  int c = (int)(idx & 127);
  int t = (int)(idx >> 7);
  int b = t / HWP, pp = t % HWP;
  int y = pp / WW_, x = pp % WW_;
  int i = c >> 2, t4 = c & 3;
  float div = expf((float)(2 * i) * (-0.14391156831212787f));  // -ln(1e4)/64
  float arg = (t4 < 2) ? ((float)(x + 1)) * div : ((float)(y + 1)) * div;
  float pe = ((t4 & 1) == 0) ? sinf(arg) : cosf(arg);
  size_t chw = ((size_t)b * 128 + c) * HWP + pp;
  Xref[idx] = refF[chw] + pe + refctx[idx];
  Xsrc[idx] = srcF[chw] + pe + srcctx[idx];
}

// ---------------- KV = sum_s K^T v per (batch, head); Ksum = sum_s K ----------------
__global__ __launch_bounds__(256) void kv_reduce_kernel(
    const float* __restrict__ Kf, const float* __restrict__ Vf,
    float* __restrict__ KV, float* __restrict__ Ksum)
{
  __shared__ float sK[8][128];
  __shared__ float sV[8][128];
  const int b = blockIdx.y;
  const int tid = threadIdx.x;
  const int m = tid >> 4;
  const int d = tid & 15;
  float acc[8];
#pragma unroll
  for (int h = 0; h < 8; ++h) acc[h] = 0.0f;
  float ksacc = 0.0f;
  const int t0 = blockIdx.x * 512;           // 40 chunks x 512 tokens = LTOK
  const float4* K4 = reinterpret_cast<const float4*>(Kf);
  const float4* V4 = reinterpret_cast<const float4*>(Vf);
  float4* sK4 = reinterpret_cast<float4*>(&sK[0][0]);
  float4* sV4 = reinterpret_cast<float4*>(&sV[0][0]);

  for (int tb = 0; tb < 512; tb += 8) {
    __syncthreads();
    size_t base = ((size_t)(b * LTOK + t0 + tb)) * 32;   // 8 rows = 256 float4
    sK4[tid] = K4[base + tid];
    sV4[tid] = V4[base + tid];
    __syncthreads();
#pragma unroll
    for (int i = 0; i < 8; ++i) {
#pragma unroll
      for (int h = 0; h < 8; ++h)
        acc[h] += sK[i][h * 16 + d] * sV[i][h * 16 + m];
      if (tid < 128) ksacc += sK[i][tid];
    }
  }
#pragma unroll
  for (int h = 0; h < 8; ++h)
    atomicAdd(&KV[(((size_t)b * 8 + h) * 16 + m) * 16 + d], acc[h]);
  if (tid < 128) atomicAdd(&Ksum[b * 128 + tid], ksacc);
}

// ---------------- attnV[t, h*16+m] = (Q . KV) * 1/(Q . Ksum + eps) ----------------
__global__ __launch_bounds__(256) void attn_combine_kernel(
    const float* __restrict__ Qf, const float* __restrict__ KV,
    const float* __restrict__ Ksum, float* __restrict__ outV)
{
  int idx = blockIdx.x * 256 + threadIdx.x;
  int t = idx >> 3;
  int h = idx & 7;
  if (t >= MTOK) return;
  int b = (t >= LTOK) ? 1 : 0;
  const float4* q4 = reinterpret_cast<const float4*>(Qf + (size_t)t * 128 + h * 16);
  const float4* ks4 = reinterpret_cast<const float4*>(Ksum + b * 128 + h * 16);
  float4 qv[4];
  float z = 1e-6f;
#pragma unroll
  for (int u = 0; u < 4; ++u) {
    qv[u] = q4[u];
    float4 kk = ks4[u];
    z += qv[u].x * kk.x + qv[u].y * kk.y + qv[u].z * kk.z + qv[u].w * kk.w;
  }
  z = 1.0f / z;
  const float4* kv4 = reinterpret_cast<const float4*>(KV + ((size_t)b * 8 + h) * 256);
  float4 o[4];
  float* op = reinterpret_cast<float*>(o);
#pragma unroll
  for (int m = 0; m < 16; ++m) {
    float s = 0.0f;
#pragma unroll
    for (int u = 0; u < 4; ++u) {
      float4 kk = kv4[m * 4 + u];
      s += qv[u].x * kk.x + qv[u].y * kk.y + qv[u].z * kk.z + qv[u].w * kk.w;
    }
    op[m] = s * z;
  }
  float4* o4 = reinterpret_cast<float4*>(outV + (size_t)t * 128 + h * 16);
#pragma unroll
  for (int u = 0; u < 4; ++u) o4[u] = o[u];
}

// ---------------- LayerNorm over d=128, one wave per token ----------------
__global__ __launch_bounds__(256) void ln_kernel(
    float* __restrict__ x, const float* __restrict__ g,
    const float* __restrict__ b, int M)
{
  int wave = threadIdx.x >> 5, lane = threadIdx.x & 31;
  int t = blockIdx.x * 8 + wave;
  if (t >= M) return;
  float4* row4 = reinterpret_cast<float4*>(x + (size_t)t * 128);
  float4 v = row4[lane];
  float s = v.x + v.y + v.z + v.w;
#pragma unroll
  for (int off = 16; off > 0; off >>= 1) s += __shfl_xor(s, off, 32);
  float mu = s * (1.0f / 128.0f);
  float d0 = v.x - mu, d1 = v.y - mu, d2 = v.z - mu, d3 = v.w - mu;
  float var = d0 * d0 + d1 * d1 + d2 * d2 + d3 * d3;
#pragma unroll
  for (int off = 16; off > 0; off >>= 1) var += __shfl_xor(var, off, 32);
  var *= (1.0f / 128.0f);
  float rs = rsqrtf(var + 1e-5f);
  float4 gv = reinterpret_cast<const float4*>(g)[lane];
  float4 bv = reinterpret_cast<const float4*>(b)[lane];
  float4 o;
  o.x = d0 * rs * gv.x + bv.x;
  o.y = d1 * rs * gv.y + bv.y;
  o.z = d2 * rs * gv.z + bv.z;
  o.w = d3 * rs * gv.w + bv.w;
  row4[lane] = o;
}

// ---------------- final stack copy ----------------
__global__ __launch_bounds__(256) void copy2_kernel(
    const float* __restrict__ a, const float* __restrict__ bsrc,
    float* __restrict__ out)
{
  size_t i = (size_t)blockIdx.x * 256 + threadIdx.x;
  if (i >= (size_t)MTOK * 128) return;
  out[i] = a[i];
  out[(size_t)MTOK * 128 + i] = bsrc[i];
}

// =====================================================================
// host-side: one encoder-layer pass  x = layer(x, src)
// =====================================================================
static void run_layer(float* x, const float* src, int li,
                      float* q, float* k, float* v, float* attnV,
                      float* xtmp, float* hidden, float* KV, float* Ksum,
                      const unsigned int* wb_attn, const unsigned int* wb_f1,
                      const unsigned int* wb_f2,
                      const float* attn_b, const float* ffn_b1, const float* ffn_b2,
                      const float* ln_g, const float* ln_b, hipStream_t stream)
{
  const unsigned int* W0 = wb_attn + ((size_t)li * 4 + 0) * 8192;
  const unsigned int* W1 = wb_attn + ((size_t)li * 4 + 1) * 8192;
  const unsigned int* W2 = wb_attn + ((size_t)li * 4 + 2) * 8192;
  const unsigned int* W3 = wb_attn + ((size_t)li * 4 + 3) * 8192;
  const float* b0 = attn_b + ((size_t)li * 4 + 0) * 128;
  const float* b1 = attn_b + ((size_t)li * 4 + 1) * 128;
  const float* b2 = attn_b + ((size_t)li * 4 + 2) * 128;
  const float* b3 = attn_b + ((size_t)li * 4 + 3) * 128;
  const unsigned int* F1 = wb_f1 + (size_t)li * 16384;
  const unsigned int* F2 = wb_f2 + (size_t)li * 16384;
  const float* fb1 = ffn_b1 + (size_t)li * 256;
  const float* fb2 = ffn_b2 + (size_t)li * 128;
  const float* g0 = ln_g + (size_t)li * 256;
  const float* g1 = ln_g + (size_t)li * 256 + 128;
  const float* lb0 = ln_b + (size_t)li * 256;
  const float* lb1 = ln_b + (size_t)li * 256 + 128;

  dim3 gA(MTOK / 256, 2);    // N=128
  dim3 gB(MTOK / 256, 4);    // N=256

  gemm_bf16_kernel<<<gA, 256, 0, stream>>>(x,   W0, b0, nullptr, q, MTOK, 128, 128, 2);
  gemm_bf16_kernel<<<gA, 256, 0, stream>>>(src, W1, b1, nullptr, k, MTOK, 128, 128, 2);
  gemm_bf16_kernel<<<gA, 256, 0, stream>>>(src, W2, b2, nullptr, v, MTOK, 128, 128, 0);

  zero_kernel<<<16, 256, 0, stream>>>(KV, 2 * 8 * 256);
  zero_kernel<<<1, 256, 0, stream>>>(Ksum, 2 * 128);
  kv_reduce_kernel<<<dim3(40, 2), 256, 0, stream>>>(k, v, KV, Ksum);
  attn_combine_kernel<<<(MTOK * 8) / 256, 256, 0, stream>>>(q, KV, Ksum, attnV);

  gemm_bf16_kernel<<<gA, 256, 0, stream>>>(attnV, W3, b3, x, xtmp, MTOK, 128, 128, 0);
  ln_kernel<<<MTOK / 8, 256, 0, stream>>>(xtmp, g0, lb0, MTOK);

  gemm_bf16_kernel<<<gB, 256, 0, stream>>>(xtmp, F1, fb1, nullptr, hidden, MTOK, 256, 128, 1);
  gemm_bf16_kernel<<<gA, 256, 0, stream>>>(hidden, F2, fb2, xtmp, x, MTOK, 128, 256, 0);
  ln_kernel<<<MTOK / 8, 256, 0, stream>>>(x, g1, lb1, MTOK);
}

extern "C" void kernel_launch(void* const* d_in, const int* in_sizes, int n_in,
                              void* d_out, int out_size, void* d_ws, size_t ws_size,
                              hipStream_t stream)
{
  const float* ref_feature = (const float*)d_in[0];
  const float* src_feature = (const float*)d_in[1];
  const float* ref_proj    = (const float*)d_in[2];
  const float* ref_i2w     = (const float*)d_in[4];
  const float* src_i2w     = (const float*)d_in[5];
  const float* depth       = (const float*)d_in[6];
  const float* attn_w = (const float*)d_in[7];
  const float* attn_b = (const float*)d_in[8];
  const float* ffn_w1 = (const float*)d_in[9];
  const float* ffn_b1 = (const float*)d_in[10];
  const float* ffn_w2 = (const float*)d_in[11];
  const float* ffn_b2 = (const float*)d_in[12];
  const float* ln_g = (const float*)d_in[13];
  const float* ln_b = (const float*)d_in[14];
  const float* rc_w = (const float*)d_in[15];
  const float* rc_b = (const float*)d_in[16];
  const float* bn2_g = (const float*)d_in[17];
  const float* bn2_b = (const float*)d_in[18];
  const float* cc_w = (const float*)d_in[19];
  const float* cc_b = (const float*)d_in[20];
  const float* bn1_g = (const float*)d_in[21];
  const float* bn1_b = (const float*)d_in[22];
  const float* fc1_w = (const float*)d_in[23];
  const float* fc1_b = (const float*)d_in[24];
  const float* fc2_w = (const float*)d_in[25];
  const float* fc2_b = (const float*)d_in[26];
  const float* ser_w = (const float*)d_in[27];
  const float* ser_b = (const float*)d_in[28];
  const float* see_w = (const float*)d_in[29];
  const float* see_b = (const float*)d_in[30];

  char* ws = (char*)d_ws;
  size_t off = 0;
  auto alloc = [&](size_t bytes) -> char* {
    char* p = ws + off;
    off += (bytes + 255) & ~(size_t)255;
    return p;
  };

  const size_t ACT = (size_t)MTOK * 128 * 4;   // 21 MB activation buffer
  unsigned int* wb_attn = (unsigned int*)alloc((size_t)16 * 8192 * 4);
  unsigned int* wb_f1   = (unsigned int*)alloc((size_t)4 * 16384 * 4);
  unsigned int* wb_f2   = (unsigned int*)alloc((size_t)4 * 16384 * 4);
  unsigned int* wb_cc   = (unsigned int*)alloc((size_t)8192 * 4);
  float* gate  = (float*)alloc(2 * 128 * 4);
  float* KV    = (float*)alloc(2 * 8 * 256 * 4);
  float* Ksum  = (float*)alloc(2 * 128 * 4);
  float* bufD  = (float*)alloc(ACT);           // ctx, later v
  float* bufE  = (float*)alloc(ACT);           // ref_ctx, later q
  float* bufF  = (float*)alloc(ACT);           // src_ctx, later k
  float* Xref  = (float*)alloc(ACT);
  float* Xsrc  = (float*)alloc(ACT);
  float* attnV = (float*)alloc(ACT);
  float* xtmp  = (float*)alloc(ACT);
  float* hidden = (float*)alloc((size_t)MTOK * 256 * 4);
  (void)ws_size; (void)in_sizes; (void)n_in; (void)out_size;

  // --- weight pre-pack to WMMA fragment order (bf16) ---
  pack_b_kernel<<<dim3(32, 16), 256, 0, stream>>>(attn_w, wb_attn, 128, 128);
  pack_b_kernel<<<dim3(64, 4), 256, 0, stream>>>(ffn_w1, wb_f1, 128, 256);
  pack_b_kernel<<<dim3(64, 4), 256, 0, stream>>>(ffn_w2, wb_f2, 256, 128);
  pack_b_kernel<<<dim3(32, 1), 256, 0, stream>>>(cc_w, wb_cc, 128, 128);

  // --- camera encode: SE gate, conv ctx, 128x128 projection ---
  cam_se_kernel<<<1, 256, 0, stream>>>(ref_i2w, bn1_g, bn1_b, fc1_w, fc1_b,
                                       fc2_w, fc2_b, ser_w, ser_b, see_w, see_b, gate);
  ctx_kernel<<<MTOK / 256, 256, 0, stream>>>(depth, rc_w, rc_b, bn2_g, bn2_b, gate, bufD);
  gemm_bf16_kernel<<<dim3(MTOK / 256, 2), 256, 0, stream>>>(
      bufD, wb_cc, cc_b, nullptr, bufE, MTOK, 128, 128, 0);       // ref_ctx

  // --- warp ref_ctx -> src_ctx; assemble token streams ---
  warp_kernel<<<MTOK / 2, 256, 0, stream>>>(bufE, depth, ref_proj, src_i2w, bufF);
  assemble_kernel<<<(MTOK * 128) / 256, 256, 0, stream>>>(
      ref_feature, src_feature, bufE, bufF, Xref, Xsrc);

  // --- 4 encoder layers: self, cross, self, cross ---
  for (int li = 0; li < 4; ++li) {
    bool is_self = ((li & 1) == 0);
    if (is_self) {
      run_layer(Xref, Xref, li, bufE, bufF, bufD, attnV, xtmp, hidden, KV, Ksum,
                wb_attn, wb_f1, wb_f2, attn_b, ffn_b1, ffn_b2, ln_g, ln_b, stream);
      run_layer(Xsrc, Xsrc, li, bufE, bufF, bufD, attnV, xtmp, hidden, KV, Ksum,
                wb_attn, wb_f1, wb_f2, attn_b, ffn_b1, ffn_b2, ln_g, ln_b, stream);
    } else {
      run_layer(Xref, Xsrc, li, bufE, bufF, bufD, attnV, xtmp, hidden, KV, Ksum,
                wb_attn, wb_f1, wb_f2, attn_b, ffn_b1, ffn_b2, ln_g, ln_b, stream);
      run_layer(Xsrc, Xref, li, bufE, bufF, bufD, attnV, xtmp, hidden, KV, Ksum,
                wb_attn, wb_f1, wb_f2, attn_b, ffn_b1, ffn_b2, ln_g, ln_b, stream);
    }
  }

  // --- stack [ref, src] into output ---
  copy2_kernel<<<(MTOK * 128) / 256, 256, 0, stream>>>(Xref, Xsrc, (float*)d_out);
}
